// Quant_Conv2d_38182259261663
// MI455X (gfx1250) — 8-bit matrix operand order repaired and checked by reading, compiled, not run on hardware
//
#include <hip/hip_runtime.h>
#include <hip/hip_bf16.h>
#include <stdint.h>


typedef int v8i __attribute__((ext_vector_type(8)));
typedef int v4i_vs __attribute__((vector_size(16)));

#define AS1 __attribute__((address_space(1)))
#define AS3 __attribute__((address_space(3)))

#define N_IMG   32
#define CH      64
#define OCH     64
#define HW      112
#define HWHW    (HW * HW)
#define OUT_ELEMS (25690112ull)
#define XPACK_BYTES (25690112ull)
#define WPACK_BYTES (9 * 64 * 64)
#define ROW_BYTES (HW * 64)

#if __has_builtin(__builtin_amdgcn_global_load_async_to_lds_b128)
#define HAVE_ASYNC_LDS 1
#else
#define HAVE_ASYNC_LDS 0
#endif

__global__ __launch_bounds__(64) void quant_w_kernel(
    const float* __restrict__ weight,
    const float* __restrict__ scale_x,
    const float* __restrict__ bias,
    signed char* __restrict__ wpack,
    int* __restrict__ bq,
    float* __restrict__ scale_out)
{
    __shared__ float smax[64];
    const int o = blockIdx.x;
    const int c = threadIdx.x;

    float wl[9];
    float m = 0.0f;
#pragma unroll
    for (int t = 0; t < 9; ++t) {
        wl[t] = weight[(o * 64 + c) * 9 + t];
        m = fmaxf(m, fabsf(wl[t]));
    }
    smax[c] = m;
    __syncthreads();
#pragma unroll
    for (int s = 32; s > 0; s >>= 1) {
        if (c < s) smax[c] = fmaxf(smax[c], smax[c + s]);
        __syncthreads();
    }
    const float scale_w = fmaxf(smax[0], 1e-8f) / 127.0f;

#pragma unroll
    for (int t = 0; t < 9; ++t) {
        float q = rintf(wl[t] / scale_w);
        q = fminf(fmaxf(q, -128.0f), 127.0f);
        wpack[(t * 64 + o) * 64 + c] = (signed char)(int)q;
    }
    if (c == 0) {
        const float so = scale_w * scale_x[0];
        scale_out[o] = so;
        float bf = rintf(bias[o] / so);
        bf = fminf(fmaxf(bf, -2147483648.0f), 2147483520.0f);
        bq[o] = (int)bf;
    }
}

__global__ __launch_bounds__(256) void pack_x_kernel(
    const int* __restrict__ xq,
    unsigned char* __restrict__ xpack)
{
    const size_t idx = (size_t)blockIdx.x * 256 + threadIdx.x;
    const int w  = (int)(idx % HW);
    const int h  = (int)((idx / HW) % HW);
    const int c4 = (int)((idx / HWHW) % 16);
    const int n  = (int)(idx / ((size_t)HWHW * 16));

    unsigned int v = 0;
#pragma unroll
    for (int j = 0; j < 4; ++j) {
        const size_t src = (((size_t)n * CH + (c4 * 4 + j)) * HW + h) * HW + w;
        v |= ((unsigned int)(unsigned char)xq[src]) << (8 * j);
    }
    *(unsigned int*)(xpack + ((((size_t)n * HW + h) * HW + w) << 6) + c4 * 4) = v;
}

template <bool PACKED>
__global__ __launch_bounds__(256) void conv_wmma_kernel(
    const unsigned char* __restrict__ xpack,
    const int* __restrict__ xq,
    const signed char* __restrict__ wpack,
    const int* __restrict__ bq,
    int* __restrict__ out)
{
    __shared__ __align__(16) unsigned char lds_x[4 * ROW_BYTES];

    const int lane = threadIdx.x & 31;
    const int wv   = threadIdx.x >> 5;
    const int blk  = blockIdx.x;
    const int y0   = (blk % 56) * 2;
    const int n    = blk / 56;
    const int ot   = wv & 3;
    const int y    = y0 + (wv >> 2);

    const int half = lane >> 4;
    const int m    = lane & 15;

    if (PACKED) {
        const int tid = threadIdx.x;
#pragma unroll
        for (int i = 0; i < 7; ++i) {
            const int lin   = i * 256 + tid;
            const int row   = lin / 448;
            const int chunk = lin - row * 448;
            const int yy    = y0 - 1 + row;
            if (yy >= 0 && yy < HW) {
                const unsigned char* g = xpack +
                    ((((size_t)n * HW + yy) * HW) << 6) + chunk * 16;
                unsigned char* l = &lds_x[row * ROW_BYTES + chunk * 16];
#if HAVE_ASYNC_LDS
                __builtin_amdgcn_global_load_async_to_lds_b128(
                    (AS1 v4i_vs*)g, (AS3 v4i_vs*)l, 0, 0);
#else
                *(int4*)l = *(const int4*)g;
#endif
            }
        }
#if HAVE_ASYNC_LDS
#if __has_builtin(__builtin_amdgcn_s_wait_asynccnt)
        __builtin_amdgcn_s_wait_asynccnt(0);
#else
        asm volatile("s_wait_asynccnt 0" ::: "memory");
#endif
#endif
        __syncthreads();
    }

    v8i acc[7] = {};
    const int o_a = ot * 16 + m;

#pragma unroll
    for (int kh = 0; kh < 3; ++kh) {
        const int yy = y + kh - 1;
        if (yy < 0 || yy >= HW) continue;
        const int slot = yy - (y0 - 1);
#pragma unroll
        for (int kw = 0; kw < 3; ++kw) {
            const int t = kh * 3 + kw;
            v8i a;
            {
                const unsigned char* wp =
                    (const unsigned char*)wpack + (t * 64 + o_a) * 64 + half * 16;
#pragma unroll
                for (int g2 = 0; g2 < 4; ++g2) {
                    const int2 d = *(const int2*)(wp + (g2 >> 1) * 32 + (g2 & 1) * 8);
                    a[2 * g2]     = d.x;
                    a[2 * g2 + 1] = d.y;
                }
            }
#pragma unroll
            for (int wt = 0; wt < 7; ++wt) {
                const int xx = wt * 16 + m + kw - 1;
                const bool xv = (xx >= 0) && (xx < HW);
                const int xc = xv ? xx : 0;
                v8i b;
                if (PACKED) {
                    const int base = slot * ROW_BYTES + xc * 64 + half * 16;
                    const int4 lo = *(const int4*)&lds_x[base];
                    const int4 hi = *(const int4*)&lds_x[base + 32];
                    b[0] = xv ? lo.x : 0;  b[1] = xv ? lo.y : 0;
                    b[2] = xv ? lo.z : 0;  b[3] = xv ? lo.w : 0;
                    b[4] = xv ? hi.x : 0;  b[5] = xv ? hi.y : 0;
                    b[6] = xv ? hi.z : 0;  b[7] = xv ? hi.w : 0;
                } else {
#pragma unroll
                    for (int v = 0; v < 8; ++v) {
                        const int cbase = ((v >> 2) << 5) + (half << 4) + ((v & 3) << 2);
                        unsigned int val = 0;
#pragma unroll
                        for (int j = 0; j < 4; ++j) {
                            const size_t src =
                                (((size_t)n * CH + (cbase + j)) * HW + yy) * HW + xc;
                            val |= ((unsigned int)(unsigned char)xq[src]) << (8 * j);
                        }
                        b[v] = xv ? (int)val : 0;
                    }
                }
                acc[wt] = __builtin_amdgcn_wmma_i32_16x16x64_iu8(
                     true, a,  false, b, acc[wt],
                     false,  false);
            }
        }
    }

    int bsum[8];
#pragma unroll
    for (int r = 0; r < 8; ++r) bsum[r] = bq[ot * 16 + half * 8 + r];

#pragma unroll
    for (int wt = 0; wt < 7; ++wt) {
        const int w = wt * 16 + m;
#pragma unroll
        for (int r = 0; r < 8; ++r) {
            const int oo = ot * 16 + half * 8 + r;
            out[(((size_t)n * OCH + oo) * HW + y) * HW + w] = acc[wt][r] + bsum[r];
        }
    }
}

extern "C" void kernel_launch(void* const* d_in, const int* in_sizes, int n_in,
                              void* d_out, int out_size, void* d_ws, size_t ws_size,
                              hipStream_t stream) {
    const int*   xq      = (const int*)d_in[0];
    const float* scale_x = (const float*)d_in[1];
    const float* weight  = (const float*)d_in[2];
    const float* bias    = (const float*)d_in[3];

    int*   outq      = (int*)d_out;
    float* scale_out = (float*)d_out + OUT_ELEMS;

    unsigned char* ws = (unsigned char*)d_ws;
    const bool packed = ws_size >= XPACK_BYTES + WPACK_BYTES + 256;

    unsigned char* xpack;
    signed char*   wpack;
    int*           bq;
    if (packed) {
        xpack = ws;
        wpack = (signed char*)(ws + XPACK_BYTES);
        bq    = (int*)(ws + XPACK_BYTES + WPACK_BYTES);
    } else {
        xpack = nullptr;
        wpack = (signed char*)ws;
        bq    = (int*)(ws + WPACK_BYTES);
    }

    quant_w_kernel<<<64, 64, 0, stream>>>(weight, scale_x, bias, wpack, bq, scale_out);

    if (packed) {
        pack_x_kernel<<<25088, 256, 0, stream>>>(xq, xpack);
        conv_wmma_kernel<true><<<1792, 256, 0, stream>>>(xpack, xq, wpack, bq, outq);
    } else {
        conv_wmma_kernel<false><<<1792, 256, 0, stream>>>(xpack, xq, wpack, bq, outq);
    }
}
